// LabelWiseAttention_22342419874354
// MI455X (gfx1250) — matrix operand order repaired and checked by reading, not compiled, not run on hardware
//
#include <hip/hip_runtime.h>
#include <stdint.h>

// Problem constants (from reference)
#define BB 16
#define TT 4096
#define DD 512
#define LL 1024

#define L_TILE 64              // labels per block
#define T_TILE 256             // rows per block-iteration (8 waves x 32 rows)
#define NCHUNK 8               // T split across blocks
#define TCHUNK (TT / NCHUNK)   // 512 -> 2 tile iterations per block
#define NPART  (NCHUNK * 8)    // partials per (b,l): 8 chunks x 8 waves = 64

typedef __attribute__((ext_vector_type(16))) __bf16 v16bf;
typedef __attribute__((ext_vector_type(8)))  float  v8f;

union FragBF { uint4 u[2]; v16bf v; };

static __device__ __forceinline__ uint16_t f2bf(float f) {
    uint32_t u = __float_as_uint(f);
    uint32_t r = u + 0x7FFFu + ((u >> 16) & 1u);   // round-to-nearest-even
    return (uint16_t)(r >> 16);
}

// ---------------------------------------------------------------- f32 -> bf16
__global__ void cvt_f32_bf16_kernel(const float* __restrict__ src,
                                    uint16_t* __restrict__ dst, int n4) {
    int i = blockIdx.x * blockDim.x + threadIdx.x;
    if (i >= n4) return;
    float4 f = reinterpret_cast<const float4*>(src)[i];
    uint32_t lo = (uint32_t)f2bf(f.x) | ((uint32_t)f2bf(f.y) << 16);
    uint32_t hi = (uint32_t)f2bf(f.z) | ((uint32_t)f2bf(f.w) << 16);
    reinterpret_cast<uint2*>(dst)[i] = make_uint2(lo, hi);
}

// ------------------------------------------------- s[b,t] = dot(x[b,t,:], w)
__global__ void row_dot_kernel(const float* __restrict__ x,
                               const float* __restrict__ w,
                               float* __restrict__ s, int nrows) {
    int wave = threadIdx.x >> 5;
    int lane = threadIdx.x & 31;
    int row  = blockIdx.x * 8 + wave;
    if (row >= nrows) return;
    const float* xr = x + (size_t)row * DD;
    float acc = 0.f;
    #pragma unroll
    for (int i = 0; i < DD / 32; ++i) acc += xr[lane + i * 32] * w[lane + i * 32];
    #pragma unroll
    for (int off = 16; off > 0; off >>= 1) acc += __shfl_xor(acc, off, 32);
    if (lane == 0) s[row] = acc;
}

// ------------------------------------------------------- fused attention core
// grid: x = (L/L_TILE)*NCHUNK, y = B ; block = 256 (8 waves, 32 rows x 64 cols each)
__global__ __launch_bounds__(256)
void attn_kernel(const uint16_t* __restrict__ xbf,
                 const uint16_t* __restrict__ qbf,
                 const float* __restrict__ s,
                 float* __restrict__ pm, float* __restrict__ pd,
                 float* __restrict__ pn) {
    __shared__ uint4 qt[(L_TILE * DD * 2) / 16];   // 64 KB: 64 labels x 512 bf16

    const int ltile = blockIdx.x % (LL / L_TILE);
    const int chunk = blockIdx.x / (LL / L_TILE);
    const int b     = blockIdx.y;
    const int l0    = ltile * L_TILE;
    const int wave  = threadIdx.x >> 5;
    const int lane  = threadIdx.x & 31;
    const int lhalf = lane >> 4;                   // 0: lanes 0-15, 1: lanes 16-31
    const int lmod  = lane & 15;

    // stage q tile (contiguous 64 KB region of row-major q_bf16) into LDS
    {
        const uint4* src = reinterpret_cast<const uint4*>(
            reinterpret_cast<const char*>(qbf) + (size_t)l0 * DD * 2);
        for (int i = threadIdx.x; i < (L_TILE * DD * 2) / 16; i += 256)
            qt[i] = src[i];
    }
    __syncthreads();

    float m_s[4], d_s[4], n_s[4];
    #pragma unroll
    for (int j = 0; j < 4; ++j) { m_s[j] = -INFINITY; d_s[j] = 0.f; n_s[j] = 0.f; }

    const char* xb = reinterpret_cast<const char*>(xbf);
    const int t_beg = chunk * TCHUNK;
    const int t_end = t_beg + TCHUNK;

    for (int tt = t_beg; tt < t_end; tt += T_TILE) {
        const int trow0 = tt + wave * 32;          // this wave's 32 rows
        // A-fragment per-lane base (ISA 16-bit A 16x32 layout):
        //   row = trow0 + lmod (+16 for second row-tile), K-halves split by lane half
        const size_t a_base0 =
            ((size_t)(b * TT + trow0 + lmod)) * (DD * 2) + (size_t)lhalf * 16;
        const size_t a_base1 = a_base0 + (size_t)16 * DD * 2;

        if (tt + T_TILE < t_end)                   // prefetch next tile's rows
            __builtin_prefetch(xb + a_base0 + (size_t)T_TILE * DD * 2, 0, 0);

        v8f acc0[4] = {};
        v8f acc1[4] = {};
        #pragma unroll 4
        for (int k0 = 0; k0 < DD; k0 += 32) {
            // ---- gather all fragments for this K-step first (one load clause)
            FragBF a0, a1;
            const uint4* pA0 = reinterpret_cast<const uint4*>(xb + a_base0 + k0 * 2);
            const uint4* pA1 = reinterpret_cast<const uint4*>(xb + a_base1 + k0 * 2);
            a0.u[0] = pA0[0]; a0.u[1] = pA0[2];    // K 0..7|8..15 , 16..23|24..31
            a1.u[0] = pA1[0]; a1.u[1] = pA1[2];
            FragBF bq[4];
            #pragma unroll
            for (int j = 0; j < 4; ++j) {
                int idx4 = ((j * 16 + lmod) * (DD * 2) + lhalf * 16 + k0 * 2) >> 4;
                bq[j].u[0] = qt[idx4];             // K 0..7|8..15 , 16..23|24..31
                bq[j].u[1] = qt[idx4 + 2];
            }
            // ---- 8 back-to-back WMMAs (each B fragment feeds 2 matrix ops)
            #pragma unroll
            for (int j = 0; j < 4; ++j) {
                acc0[j] = __builtin_amdgcn_wmma_f32_16x16x32_bf16(
                    false, a0.v, false, bq[j].v, (short)0, acc0[j], false, false);
                acc1[j] = __builtin_amdgcn_wmma_f32_16x16x32_bf16(
                    false, a1.v, false, bq[j].v, (short)0, acc1[j], false, false);
            }
        }

        // per-accumulator-row s values: global row = trow0 + rt*16 + r + 8*lhalf
        float sv0[8], sv1[8];
        const float* srow = s + (size_t)b * TT + trow0 + lhalf * 8;
        #pragma unroll
        for (int r = 0; r < 8; ++r) { sv0[r] = srow[r]; sv1[r] = srow[16 + r]; }

        // online softmax update per label column (column n lives in lanes n, n+16)
        #pragma unroll
        for (int j = 0; j < 4; ++j) {
            float tmax = acc0[j][0];
            #pragma unroll
            for (int r = 1; r < 8; ++r) tmax = fmaxf(tmax, acc0[j][r]);
            #pragma unroll
            for (int r = 0; r < 8; ++r) tmax = fmaxf(tmax, acc1[j][r]);
            tmax = fmaxf(tmax, __shfl_xor(tmax, 16, 32));
            float m_new = fmaxf(m_s[j], tmax);
            float ps = 0.f, pw = 0.f;
            #pragma unroll
            for (int r = 0; r < 8; ++r) {
                float e0 = __expf(acc0[j][r] - m_new);
                float e1 = __expf(acc1[j][r] - m_new);
                ps += e0 + e1;
                pw += e0 * sv0[r] + e1 * sv1[r];
            }
            ps += __shfl_xor(ps, 16, 32);
            pw += __shfl_xor(pw, 16, 32);
            float scale = __expf(m_s[j] - m_new);  // 0 on first tile (m=-inf)
            d_s[j] = d_s[j] * scale + ps;
            n_s[j] = n_s[j] * scale + pw;
            m_s[j] = m_new;
        }
    }

    if (lane < 16) {
        int p = chunk * 8 + wave;
        #pragma unroll
        for (int j = 0; j < 4; ++j) {
            size_t idx = ((size_t)b * NPART + p) * LL + (l0 + j * 16 + lane);
            pm[idx] = m_s[j];
            pd[idx] = d_s[j];
            pn[idx] = n_s[j];
        }
    }
}

// ---------------------------------------------------------------- merge pass
__global__ void merge_kernel(const float* __restrict__ pm,
                             const float* __restrict__ pd,
                             const float* __restrict__ pn,
                             const float* __restrict__ fcb,
                             float* __restrict__ out) {
    int gid = blockIdx.x * blockDim.x + threadIdx.x;
    if (gid >= BB * LL) return;
    int b = gid / LL, l = gid % LL;
    size_t base = (size_t)b * NPART * LL + l;
    float M = -INFINITY;
    for (int p = 0; p < NPART; ++p) M = fmaxf(M, pm[base + (size_t)p * LL]);
    float den = 0.f, num = 0.f;
    for (int p = 0; p < NPART; ++p) {
        float w = __expf(pm[base + (size_t)p * LL] - M);
        den += w * pd[base + (size_t)p * LL];
        num += w * pn[base + (size_t)p * LL];
    }
    out[gid] = num / den + fcb[0];
}

// ---------------------------------------------------------------------- host
extern "C" void kernel_launch(void* const* d_in, const int* in_sizes, int n_in,
                              void* d_out, int out_size, void* d_ws, size_t ws_size,
                              hipStream_t stream) {
    (void)in_sizes; (void)n_in; (void)out_size; (void)ws_size;
    const float* x    = (const float*)d_in[0];   // (B,T,D)
    const float* q    = (const float*)d_in[1];   // (L,D)
    const float* fc_w = (const float*)d_in[2];   // (1,D)
    const float* fc_b = (const float*)d_in[3];   // (1,)
    float* out = (float*)d_out;                  // (B,L)

    // workspace layout
    char* ws = (char*)d_ws;
    uint16_t* xbf = (uint16_t*)ws;                               // 64 MB
    uint16_t* qbf = (uint16_t*)(ws + (size_t)BB * TT * DD * 2);  // 1 MB
    float*    sA  = (float*)((char*)qbf + (size_t)LL * DD * 2);  // 256 KB
    float*    pm  = (float*)((char*)sA + (size_t)BB * TT * 4);
    float*    pd  = pm + (size_t)BB * NPART * LL;
    float*    pn  = pd + (size_t)BB * NPART * LL;

    {   // x -> bf16
        int n4 = BB * TT * DD / 4;
        cvt_f32_bf16_kernel<<<(n4 + 255) / 256, 256, 0, stream>>>(x, xbf, n4);
    }
    {   // q -> bf16
        int n4 = LL * DD / 4;
        cvt_f32_bf16_kernel<<<(n4 + 255) / 256, 256, 0, stream>>>(q, qbf, n4);
    }
    {   // s[b,t] = x[b,t] . fc_w
        int nrows = BB * TT;
        row_dot_kernel<<<nrows / 8, 256, 0, stream>>>(x, fc_w, sA, nrows);
    }
    {   // fused GEMM + online softmax partials
        dim3 grid((LL / L_TILE) * NCHUNK, BB);
        attn_kernel<<<grid, 256, 0, stream>>>(xbf, qbf, sA, pm, pd, pn);
    }
    {   // merge partials + bias
        merge_kernel<<<(BB * LL + 255) / 256, 256, 0, stream>>>(pm, pd, pn, fc_b, out);
    }
}
